// QuantumNeuralNetwork_23648089931809
// MI455X (gfx1250) — matrix operand order repaired and checked by reading, not compiled, not run on hardware
//
#include <hip/hip_runtime.h>
#include <math.h>

typedef __attribute__((ext_vector_type(16))) _Float16 v16h;
typedef __attribute__((ext_vector_type(8)))  float    v8f;
typedef int v4i __attribute__((__vector_size__(16)));   // matches async builtin param type

#define NQ 7
#define HD 128          // Hilbert dim = 2^7
#define NL 2
#define BM 64           // batch rows per tile
#define WCOLS 256       // 128 Re + 128 Im output columns
#define GRIDMAX 512     // persistent blocks (~2 WGs/WGP at 152KB LDS)

#define AS1 __attribute__((address_space(1)))
#define AS3 __attribute__((address_space(3)))

#if __has_builtin(__builtin_amdgcn_global_load_async_to_lds_b128)
#define HAS_ASYNC_LDS 1
#endif

__device__ inline void wait_async0() {
#ifdef HAS_ASYNC_LDS
#if __has_builtin(__builtin_amdgcn_s_wait_asynccnt)
    __builtin_amdgcn_s_wait_asynccnt(0);
#else
    asm volatile("s_wait_asynccnt 0x0" ::: "memory");
#endif
#endif
}

__device__ inline void async_cp16(void* lds_dst, const void* gsrc) {
#ifdef HAS_ASYNC_LDS
    __builtin_amdgcn_global_load_async_to_lds_b128(
        (AS1 v4i*)(void*)gsrc, (AS3 v4i*)lds_dst, 0, 0);
#else
    *(float4*)lds_dst = *(const float4*)gsrc;
#endif
}

struct cplx { float re, im; };
__device__ inline cplx cmul(cplx a, cplx b){ return {a.re*b.re - a.im*b.im, a.re*b.im + a.im*b.re}; }
__device__ inline cplx cadd(cplx a, cplx b){ return {a.re + b.re, a.im + b.im}; }

// ---------------------------------------------------------------------------
// Kernel 1: build the 128x128 circuit unitary in LDS, emit f16 W[256][128]
// W[n][j]: n<128 -> Re U[n][j], n>=128 -> Im U[n-128][j]  (j = K index)
// ---------------------------------------------------------------------------
__global__ __launch_bounds__(256) void build_unitary(const float* __restrict__ w,
                                                     _Float16* __restrict__ Wout)
{
    extern __shared__ float smem1[];
    float* Ure = smem1;            // 128*128 f32
    float* Uim = smem1 + HD * HD;  // 128*128 f32
    const int tid = threadIdx.x;

    for (int idx = tid; idx < HD * HD; idx += 256) {
        int r = idx >> 7, c = idx & 127;
        Ure[idx] = (r == c) ? 1.f : 0.f;
        Uim[idx] = 0.f;
    }
    __syncthreads();

    for (int layer = 0; layer < NL; ++layer) {
        // ---- rotations: fused G = Rz(t2) * Ry(t1) * Rx(t0), wire q -> bit (6-q)
        for (int q = 0; q < NQ; ++q) {
            const int bit = 6 - q;
            const float t0 = w[(layer * NQ + q) * 3 + 0];
            const float t1 = w[(layer * NQ + q) * 3 + 1];
            const float t2 = w[(layer * NQ + q) * 3 + 2];
            const float c0 = cosf(0.5f * t0), s0 = sinf(0.5f * t0);
            const float c1 = cosf(0.5f * t1), s1 = sinf(0.5f * t1);
            const float c2 = cosf(0.5f * t2), s2 = sinf(0.5f * t2);
            cplx a00{c0, 0.f}, a01{0.f, -s0}, a10{0.f, -s0}, a11{c0, 0.f};     // Rx
            cplx b00{c1, 0.f}, b01{-s1, 0.f}, b10{s1, 0.f}, b11{c1, 0.f};      // Ry
            cplx m00 = cadd(cmul(b00, a00), cmul(b01, a10));
            cplx m01 = cadd(cmul(b00, a01), cmul(b01, a11));
            cplx m10 = cadd(cmul(b10, a00), cmul(b11, a10));
            cplx m11 = cadd(cmul(b10, a01), cmul(b11, a11));
            cplx z0{c2, -s2}, z1{c2, s2};                                      // Rz
            cplx g00 = cmul(z0, m00), g01 = cmul(z0, m01);
            cplx g10 = cmul(z1, m10), g11 = cmul(z1, m11);

            for (int t = tid; t < 64 * HD; t += 256) {
                int col = t & 127;
                int pr  = t >> 7;
                int i0  = ((pr >> bit) << (bit + 1)) | (pr & ((1 << bit) - 1));
                int i1  = i0 | (1 << bit);
                cplx A{Ure[i0 * HD + col], Uim[i0 * HD + col]};
                cplx Bv{Ure[i1 * HD + col], Uim[i1 * HD + col]};
                cplx n0 = cadd(cmul(g00, A), cmul(g01, Bv));
                cplx n1 = cadd(cmul(g10, A), cmul(g11, Bv));
                Ure[i0 * HD + col] = n0.re; Uim[i0 * HD + col] = n0.im;
                Ure[i1 * HD + col] = n1.re; Uim[i1 * HD + col] = n1.im;
            }
            __syncthreads();
        }
        // ---- ring CNOTs: swap rows with control=1, target flipped
        for (int q = 0; q < NQ; ++q) {
            const int cbit = 6 - q;
            const int tbit = 6 - ((q + 1) % NQ);
            const int lo = (cbit < tbit) ? cbit : tbit;
            const int hi = (cbit < tbit) ? tbit : cbit;
            for (int t = tid; t < 32 * HD; t += 256) {
                int col = t & 127;
                int r5  = t >> 7;
                int xi = ((r5 >> lo) << (lo + 1)) | (r5 & ((1 << lo) - 1));
                int i  = ((xi >> hi) << (hi + 1)) | (xi & ((1 << hi) - 1));
                i |= (1 << cbit);
                int j = i | (1 << tbit);
                float tr = Ure[i * HD + col]; Ure[i * HD + col] = Ure[j * HD + col]; Ure[j * HD + col] = tr;
                float ti = Uim[i * HD + col]; Uim[i * HD + col] = Uim[j * HD + col]; Uim[j * HD + col] = ti;
            }
            __syncthreads();
        }
    }

    for (int idx = tid; idx < WCOLS * HD; idx += 256) {
        int n = idx >> 7, j = idx & 127;
        float v = (n < HD) ? Ure[n * HD + j] : Uim[(n - HD) * HD + j];
        Wout[idx] = (_Float16)v;
    }
}

// ---------------------------------------------------------------------------
// Kernel 2 (persistent): per tile, C[64,256] = x̂[64,128] x W[128,256] (WMMA),
// then |psi|^2 -> 7 signed reductions -> out. W staged once per block.
// Wave tile: 2 M-tiles x 4 N-tiles (af reused x4, bf reused x2).
// ---------------------------------------------------------------------------
__global__ __launch_bounds__(256) void qnn_gemm(const float* __restrict__ x,
                                                const _Float16* __restrict__ Wg,
                                                float* __restrict__ out,
                                                int nTiles)
{
    extern __shared__ char smem[];
    _Float16* Wlds = (_Float16*)smem;                          // [256][128] halves, 64KB (K contiguous)
    _Float16* Alds = (_Float16*)(smem + 65536);                // [64][128]  halves, 16KB
    float*    psi  = (float*)(smem + 65536 + 16384);           // [64][256]  f32,    64KB
    float*    Xf   = psi;                                      // alias: x-tile staging (32KB)
    float*    nrm  = (float*)(smem + 65536 + 16384 + 65536);   // [64] f32
    float*    zp   = (float*)(smem + 65536 + 16384 + 65536 + 256); // [256][8] partials, 8KB

    const int tid  = threadIdx.x;
    const int lane = tid & 31;
    const int wv   = tid >> 5;

    // wave tiling: mg selects 32 rows (2 M-tiles), ng selects 64 cols (4 N-tiles)
    const int mg   = wv & 1;
    const int ng   = wv >> 1;
    const int l15  = lane & 15;
    const int lhi  = (lane >> 4) & 1;   // lane[4]

    // ---- prologue: stage W once per block + first x tile (async -> LDS)
    {
        const char* wsrc = (const char*)Wg;
        char*       wdst = (char*)Wlds;
#pragma unroll
        for (int i = 0; i < 16; ++i) {                          // 64KB of W
            int off = (tid + i * 256) * 16;
            async_cp16(wdst + off, wsrc + off);
        }
        const char* xsrc = (const char*)(x + (size_t)blockIdx.x * BM * HD);
        char*       xdst = (char*)Xf;
#pragma unroll
        for (int i = 0; i < 8; ++i) {                           // 32KB x tile
            int off = (tid + i * 256) * 16;
            async_cp16(xdst + off, xsrc + off);
        }
    }

    for (int tile = blockIdx.x; tile < nTiles; tile += gridDim.x) {
        const int  row0  = tile * BM;
        const int  tnext = tile + gridDim.x;
        const bool hasNext = (tnext < nTiles);
        const char* xnext = (const char*)(x + (size_t)tnext * BM * HD);

        wait_async0();           // this wave's async copies (W + x tile) done
        __syncthreads();         // all waves' copies visible

        // row norms, 4 threads per row (partials in zp)
        {
            int r = tid >> 2, seg = tid & 3;
            float s = 0.f;
            for (int k = seg * 32; k < seg * 32 + 32; ++k) {
                float v = Xf[r * HD + k]; s += v * v;
            }
            zp[tid] = s;
        }
        __syncthreads();
        if (tid < BM) {
            float s = zp[tid * 4] + zp[tid * 4 + 1] + zp[tid * 4 + 2] + zp[tid * 4 + 3];
            nrm[tid] = 1.f / fmaxf(sqrtf(s), 1e-8f);
        }
        __syncthreads();

        for (int i = tid; i < BM * HD; i += 256) {
            int r = i >> 7;
            Alds[i] = (_Float16)(Xf[i] * nrm[r]);
        }
        __syncthreads();

        // warm L2/WGP$ for the next tile while the matrix pipe runs (1 line/thread)
        if (hasNext) __builtin_prefetch((const void*)(xnext + tid * 128), 0, 1);

        v8f acc[8];      // acc[mi*4+ni]
        v8f zero = {0.f, 0.f, 0.f, 0.f, 0.f, 0.f, 0.f, 0.f};
#pragma unroll
        for (int n = 0; n < 8; ++n) acc[n] = zero;

#pragma unroll
        for (int ks = 0; ks < 4; ++ks) {
            // A fragments (16x32 f16): lane l: M=l&15; halves 0..7 -> K=32ks+8*l[4], 8..15 -> +16
            v16h af[2];
            {
                int k0 = 32 * ks + lhi * 8;
#pragma unroll
                for (int mi = 0; mi < 2; ++mi) {
                    int mrow = 16 * (2 * mg + mi) + l15;
                    ((float4*)&af[mi])[0] = *(const float4*)(Alds + mrow * HD + k0);
                    ((float4*)&af[mi])[1] = *(const float4*)(Alds + mrow * HD + k0 + 16);
                }
            }
            // B fragments (32x16 f16): lane l: N=l&15; halves 0..7 -> K=32ks+8*l[4], 8..15 -> +16
            v16h bf[4];
            {
                int kb = 32 * ks + lhi * 8;
#pragma unroll
                for (int ni = 0; ni < 4; ++ni) {
                    int ncol = (4 * ng + ni) * 16 + l15;
                    const float4* q0 = (const float4*)(Wlds + ncol * HD + kb);
                    ((float4*)&bf[ni])[0] = q0[0];
                    ((float4*)&bf[ni])[1] = q0[2];
                }
            }
#pragma unroll
            for (int mi = 0; mi < 2; ++mi)
#pragma unroll
                for (int ni = 0; ni < 4; ++ni) {
                    acc[mi * 4 + ni] = __builtin_amdgcn_wmma_f32_16x16x32_f16(
                        false, af[mi], false, bf[ni], (short)0, acc[mi * 4 + ni], false, false);
                }
        }

        // C layout: VGPR v, lane l: M = v + 8*l[4], N = l&15
#pragma unroll
        for (int mi = 0; mi < 2; ++mi) {
#pragma unroll
            for (int ni = 0; ni < 4; ++ni) {
                int ncol = (4 * ng + ni) * 16 + l15;
#pragma unroll
                for (int v = 0; v < 8; ++v) {
                    int m = 16 * (2 * mg + mi) + v + (lhi << 3);
                    psi[m * WCOLS + ncol] = acc[mi * 4 + ni][v];
                }
            }
        }
        __syncthreads();

        // epilogue stage 1: 4 threads per row, 32 basis states each
        {
            int r = tid >> 2, seg = tid & 3;
            float z[7] = {0.f, 0.f, 0.f, 0.f, 0.f, 0.f, 0.f};
            int ibase = seg * 32;
            for (int ii = 0; ii < 32; ++ii) {
                int i = ibase + ii;
                float re = psi[r * WCOLS + i];
                float im = psi[r * WCOLS + 128 + i];
                float p  = re * re + im * im;
#pragma unroll
                for (int q = 0; q < 7; ++q)
                    z[q] += ((i >> (6 - q)) & 1) ? -p : p;
            }
#pragma unroll
            for (int q = 0; q < 7; ++q) zp[tid * 8 + q] = z[q];
        }
        __syncthreads();         // psi fully consumed -> staging buffer free

        // kick off next tile's async x load (overlaps stage 2 + next prologue)
        if (hasNext) {
            char* xdst = (char*)Xf;
#pragma unroll
            for (int i = 0; i < 8; ++i) {
                int off = (tid + i * 256) * 16;
                async_cp16(xdst + off, xnext + off);
            }
        }

        // epilogue stage 2: combine 4 partials per row
        if (tid < BM) {
#pragma unroll
            for (int q = 0; q < 7; ++q) {
                float s = zp[(tid * 4 + 0) * 8 + q] + zp[(tid * 4 + 1) * 8 + q]
                        + zp[(tid * 4 + 2) * 8 + q] + zp[(tid * 4 + 3) * 8 + q];
                out[(size_t)(row0 + tid) * 7 + q] = s;
            }
        }
        // zp re-written only after the next top-of-loop __syncthreads
    }
}

// ---------------------------------------------------------------------------
extern "C" void kernel_launch(void* const* d_in, const int* in_sizes, int n_in,
                              void* d_out, int out_size, void* d_ws, size_t ws_size,
                              hipStream_t stream)
{
    const float* x = (const float*)d_in[0];     // [B,128] f32
    const float* w = (const float*)d_in[1];     // [2,7,3] f32
    float* out     = (float*)d_out;             // [B,7]   f32
    _Float16* W    = (_Float16*)d_ws;           // 256*128 halves = 64KB scratch

    const int B      = in_sizes[0] / HD;
    const int nTiles = B / BM;
    const int grid   = (nTiles < GRIDMAX) ? nTiles : GRIDMAX;

    const size_t smem_u = (size_t)HD * HD * 2 * sizeof(float);            // 128 KB
    const size_t smem_g = 65536 + 16384 + 65536 + 256 + 8192;             // ~152 KB -> 2 WGs/WGP

    (void)hipFuncSetAttribute(reinterpret_cast<const void*>(build_unitary),
                              hipFuncAttributeMaxDynamicSharedMemorySize, (int)smem_u);
    (void)hipFuncSetAttribute(reinterpret_cast<const void*>(qnn_gemm),
                              hipFuncAttributeMaxDynamicSharedMemorySize, (int)smem_g);

    build_unitary<<<1, 256, smem_u, stream>>>(w, W);
    qnn_gemm<<<grid, 256, smem_g, stream>>>(x, W, out, nTiles);
    (void)n_in; (void)out_size; (void)ws_size;
}
